// MPNEncoder_48404281426495
// MI455X (gfx1250) — matrix operand order repaired and checked by reading, not compiled, not run on hardware
//
#include <hip/hip_runtime.h>
#include <hip/hip_bf16.h>
#include <stdint.h>

// ---------------------------------------------------------------------------
// MPN encoder for MI455X (gfx1250). All GEMMs run on v_wmma_f32_16x16x32_bf16
// (bf16 operands, f32 accumulate). Weight panels are staged into LDS with
// async global->LDS copies (ASYNCcnt) and double-buffered against the WMMA
// consumption. Gather/scatter/pointwise stays fp32 VALU.
// ---------------------------------------------------------------------------

#ifndef USE_ASYNC_LDS
#define USE_ASYNC_LDS 1
#endif

typedef __attribute__((ext_vector_type(16))) __bf16 v16bf;
typedef __attribute__((ext_vector_type(8)))  float  v8f;

union Frag16 { v16bf v; uint4 q[2]; };

static constexpr int HID     = 300;
static constexpr int A_ROWS  = 64001;   // atoms (row 0 pad)
static constexpr int B_ROWS  = 128001;  // bonds (row 0 pad)
static constexpr int MOL     = 40;
static constexpr int NMOLS   = 1600;
static constexpr int KF      = 192;     // padded feature K (133/147 -> 192, %64)
static constexpr int HP      = 320;     // padded HID (300 -> 320, %64)
static constexpr int GI_LD   = 912;     // row stride of gate buffers (900 -> 912)

static __device__ __forceinline__ uint16_t f2bf(float f) {
  uint32_t u = __float_as_uint(f);
  uint32_t r = u + 0x7FFFu + ((u >> 16) & 1u);   // round-to-nearest-even
  return (uint16_t)(r >> 16);
}

// ---------------------------------------------------------------------------
// Generic bf16 WMMA GEMM: C[M,N] = act( A[M,K] * W[N,K]^T + resid + bias )
// A: bf16 row-major lda (K % 64 == 0). W: bf16 row-major ldw, rows padded
// (zeros) to next multiple of 64. Block = 256 thr = 8 waves; wave owns a
// 16x64 output tile (4x v8f accumulators). The 64-row W panel for the
// block's N range is streamed through LDS in 64-wide K chunks, double
// buffered, via global_load_async_to_lds_b128 (ASYNCcnt).
// ---------------------------------------------------------------------------
__global__ __launch_bounds__(256)
void gemm_bf16_wmma(const uint16_t* __restrict__ A, int lda,
                    const uint16_t* __restrict__ W, int ldw,
                    float* __restrict__ C, int ldc,
                    const float* __restrict__ resid, int ldr,
                    const float* __restrict__ bias,
                    int M, int N, int K, int do_relu)
{
  __shared__ uint16_t smW[2][64 * 64];   // 2 x 8KB double buffer

  const int tid  = threadIdx.x;
  const int lane = tid & 31;
  const int wave = tid >> 5;
  const int half = lane >> 4;       // 0: lanes 0-15, 1: lanes 16-31
  const int l16  = lane & 15;

  const int mbase = (blockIdx.x * 8 + wave) * 16;
  const int nbase = blockIdx.y * 64;

  // A fragment: lanes 0-15 K={kb..kb+7, kb+16..kb+23}; lanes 16-31 shifted +8.
  // Rows are clamped (never early-return: all waves must reach the barriers
  // and WMMA requires EXEC all-ones); out-of-range rows are masked at store.
  int arow = mbase + l16; if (arow >= M) arow = M - 1;
  const uint16_t* __restrict__ Ap = A + (size_t)arow * lda + (half ? 8 : 0);

  // LDS staging: 64 rows x 64 K of W = 512 x 16B; each of 256 threads moves 2.
  const int row0 = tid >> 3;            // 0..31
  const int seg  = tid & 7;             // 0..7 (16B segments across 64 K)
  const int row1 = row0 + 32;           // 32..63
  const uint16_t* __restrict__ g0 = W + (size_t)(nbase + row0) * ldw + seg * 8;
  const uint16_t* __restrict__ g1 = W + (size_t)(nbase + row1) * ldw + seg * 8;
  const int soff0 = row0 * 64 + seg * 8;
  const int soff1 = row1 * 64 + seg * 8;

  auto stage = [&](int buf, int kc) {
#if USE_ASYNC_LDS
    uint32_t l0 = (uint32_t)(uintptr_t)&smW[buf][soff0];   // ISA: LDS addr = addr[31:0]
    uint32_t l1 = (uint32_t)(uintptr_t)&smW[buf][soff1];
    uint64_t a0 = (uint64_t)(uintptr_t)(g0 + kc);
    uint64_t a1 = (uint64_t)(uintptr_t)(g1 + kc);
    asm volatile("global_load_async_to_lds_b128 %0, %1, off" :: "v"(l0), "v"(a0) : "memory");
    asm volatile("global_load_async_to_lds_b128 %0, %1, off" :: "v"(l1), "v"(a1) : "memory");
#else
    *(uint4*)&smW[buf][soff0] = *(const uint4*)(g0 + kc);
    *(uint4*)&smW[buf][soff1] = *(const uint4*)(g1 + kc);
#endif
  };
  auto stage_sync = [&]() {
#if USE_ASYNC_LDS
    asm volatile("s_wait_asynccnt 0" ::: "memory");
#endif
    __syncthreads();
  };

  v8f acc[4] = {};

  stage(0, 0);
  stage_sync();

  const int nchunks = K >> 6;
  for (int c = 0; c < nchunks; ++c) {
    const int buf = c & 1;
    if (c + 1 < nchunks) stage(buf ^ 1, (c + 1) << 6);   // prefetch next chunk
    const int kc = c << 6;
    #pragma unroll
    for (int kk = 0; kk < 64; kk += 32) {
      Frag16 fa, fb[4];
      fa.q[0] = *(const uint4*)(Ap + kc + kk);
      fa.q[1] = *(const uint4*)(Ap + kc + kk + 16);
      __builtin_prefetch(Ap + kc + kk + 128, 0, 1);
      #pragma unroll
      for (int j = 0; j < 4; ++j) {
        // B fragment from LDS: lane holds W row (n); lanes 0-15 K={kb..kb+7, kb+16..kb+23},
        // lanes 16-31 shifted +8.
        const uint16_t* bp = &smW[buf][(j * 16 + l16) * 64 + kk + (half ? 8 : 0)];
        fb[j].q[0] = *(const uint4*)(bp);
        fb[j].q[1] = *(const uint4*)(bp + 16);
      }
      acc[0] = __builtin_amdgcn_wmma_f32_16x16x32_bf16(false, fa.v, false, fb[0].v,
                                                       (short)0, acc[0], false, false);
      acc[1] = __builtin_amdgcn_wmma_f32_16x16x32_bf16(false, fa.v, false, fb[1].v,
                                                       (short)0, acc[1], false, false);
      acc[2] = __builtin_amdgcn_wmma_f32_16x16x32_bf16(false, fa.v, false, fb[2].v,
                                                       (short)0, acc[2], false, false);
      acc[3] = __builtin_amdgcn_wmma_f32_16x16x32_bf16(false, fa.v, false, fb[3].v,
                                                       (short)0, acc[3], false, false);
    }
    stage_sync();   // next buffer resident + all waves done reading this one
  }

  // C/D layout: VGPR i -> M = mbase + i + half*8 ; N = nbase + j*16 + l16
  const int mrow0 = mbase + (half ? 8 : 0);
  #pragma unroll
  for (int j = 0; j < 4; ++j) {
    const int n = nbase + j * 16 + l16;
    if (n >= N) continue;
    const float bv = bias ? bias[n] : 0.f;
    #pragma unroll
    for (int i = 0; i < 8; ++i) {
      const int m = mrow0 + i;
      if (m >= M) break;
      float v = acc[j][i] + bv;
      if (resid) v += resid[(size_t)m * ldr + n];
      if (do_relu) v = fmaxf(v, 0.f);
      C[(size_t)m * ldc + n] = v;
    }
  }
}

// --- f32 -> bf16 repack with row/col zero-padding -------------------------
__global__ void pack_bf16(const float* __restrict__ src, int sld, int scols,
                          uint16_t* __restrict__ dst, int dld, int dcols,
                          int msrc, int mdst)
{
  int idx = blockIdx.x * blockDim.x + threadIdx.x;
  int total = mdst * dcols;
  if (idx >= total) return;
  int r = idx / dcols, c = idx - r * dcols;
  float v = (r < msrc && c < scols) ? src[(size_t)r * sld + c] : 0.f;
  dst[(size_t)r * dld + c] = f2bf(v);
}

// --- per-atom neighbor aggregation: sum(m)*max(m) -------------------------
__global__ void agg_sum_max(const float* __restrict__ mb, const int* __restrict__ a2b,
                            float* __restrict__ out, int accumulate)
{
  int idx = blockIdx.x * blockDim.x + threadIdx.x;
  if (idx >= A_ROWS * HID) return;
  int a = idx / HID, h = idx - a * HID;
  const int* nb = a2b + a * 6;
  float s = 0.f, mx = -3.4e38f;
  #pragma unroll
  for (int j = 0; j < 6; ++j) {
    float v = mb[(size_t)nb[j] * HID + h];
    s += v; mx = fmaxf(mx, v);
  }
  float g = s * mx;
  out[idx] = accumulate ? out[idx] + g : g;
}

// --- bond message pre-GEMM: X = bf16(message_atom[b2a] - message_bond[b2revb])
__global__ void bond_pre(const float* __restrict__ ma, const float* __restrict__ mb,
                         const int* __restrict__ b2a, const int* __restrict__ b2revb,
                         uint16_t* __restrict__ X)
{
  int idx = blockIdx.x * blockDim.x + threadIdx.x;
  if (idx >= B_ROWS * HP) return;
  int b = idx / HP, h = idx - b * HP;
  float v = 0.f;
  if (h < HID)
    v = ma[(size_t)b2a[b] * HID + h] - mb[(size_t)b2revb[b] * HID + h];
  X[idx] = f2bf(v);
}

// --- msg = relu(node + gru_bias) in bf16 (GRU gate GEMM input) ------------
__global__ void msg_make(const float* __restrict__ node, const float* __restrict__ bias,
                         uint16_t* __restrict__ msgb)
{
  int idx = blockIdx.x * blockDim.x + threadIdx.x;
  if (idx >= A_ROWS * HP) return;
  int a = idx / HP, h = idx - a * HP;
  float v = 0.f;
  if (h < HID) v = fmaxf(node[(size_t)a * HID + h] + bias[h], 0.f);
  msgb[idx] = f2bf(v);
}

// --- padding row head: message[0] = [msg[0], msg[0]] ----------------------
__global__ void head_row(const float* __restrict__ node, const float* __restrict__ bias,
                         uint16_t* __restrict__ msgout)
{
  int h = blockIdx.x * blockDim.x + threadIdx.x;
  if (h >= HID) return;
  uint16_t b = f2bf(fmaxf(node[h] + bias[h], 0.f));
  msgout[h] = b;
  msgout[HP + h] = b;
}

// --- h0 = max over molecule atoms of node; seed both GRU directions -------
__global__ void h0_init(const float* __restrict__ node,
                        float* __restrict__ hf, float* __restrict__ hb,
                        uint16_t* __restrict__ hbf, uint16_t* __restrict__ hbb)
{
  int idx = blockIdx.x * blockDim.x + threadIdx.x;
  if (idx >= NMOLS * HP) return;
  int b = idx / HP, h = idx - b * HP;
  if (h < HID) {
    float m = -3.4e38f;
    for (int t = 0; t < MOL; ++t)
      m = fmaxf(m, node[(size_t)(1 + b * MOL + t) * HID + h]);
    hf[(size_t)b * HID + h] = m;
    hb[(size_t)b * HID + h] = m;
    uint16_t x = f2bf(m);
    hbf[idx] = x; hbb[idx] = x;
  } else {
    hbf[idx] = 0; hbb[idx] = 0;
  }
}

__global__ void fill_zero_u32(uint32_t* __restrict__ p, long long n)
{
  long long i = (long long)blockIdx.x * blockDim.x + threadIdx.x;
  if (i < n) p[i] = 0;
}

// --- one GRU cell update for all molecules (one direction, one time step) --
__global__ void gru_step(const float* __restrict__ gi, const float* __restrict__ gh,
                         float* __restrict__ hstate, uint16_t* __restrict__ hstate_bf,
                         uint16_t* __restrict__ msgout, int t_idx, int col_off)
{
  int idx = blockIdx.x * blockDim.x + threadIdx.x;
  if (idx >= NMOLS * HID) return;
  int b = idx / HID, h = idx - b * HID;
  size_t gir = (size_t)(b * MOL + t_idx) * GI_LD;
  float ir = gi[gir + h], iz = gi[gir + HID + h], in_ = gi[gir + 2 * HID + h];
  size_t ghr = (size_t)b * GI_LD;
  float hr = gh[ghr + h], hz = gh[ghr + HID + h], hn = gh[ghr + 2 * HID + h];
  float r = 1.f / (1.f + __expf(-(ir + hr)));
  float z = 1.f / (1.f + __expf(-(iz + hz)));
  float n = tanhf(in_ + r * hn);
  float hp = hstate[idx];
  float hv = (1.f - z) * n + z * hp;
  hstate[idx] = hv;
  hstate_bf[(size_t)b * HP + h] = f2bf(hv);
  msgout[(size_t)(1 + b * MOL + t_idx) * (2 * HP) + col_off + h] = f2bf(hv);
}

// --- final mean over molecule atoms ---------------------------------------
__global__ void mol_mean(const float* __restrict__ ah, float* __restrict__ out)
{
  int idx = blockIdx.x * blockDim.x + threadIdx.x;
  if (idx >= NMOLS * HID) return;
  int mol = idx / HID, h = idx - mol * HID;
  float s = 0.f;
  for (int t = 0; t < MOL; ++t)
    s += ah[(size_t)(1 + mol * MOL + t) * HID + h];
  out[idx] = s * (1.f / (float)MOL);
}

// ---------------------------------------------------------------------------
static inline int cdiv(long long a, long long b) { return (int)((a + b - 1) / b); }

extern "C" void kernel_launch(void* const* d_in, const int* in_sizes, int n_in,
                              void* d_out, int out_size, void* d_ws, size_t ws_size,
                              hipStream_t stream)
{
  (void)in_sizes; (void)n_in; (void)out_size; (void)ws_size;

  const float* f_atoms  = (const float*)d_in[0];
  const float* f_bonds  = (const float*)d_in[1];
  const float* W_i_atom = (const float*)d_in[2];
  const float* W_i_bond = (const float*)d_in[3];
  const float* W_h      = (const float*)d_in[4];   // [5,300,300]
  const float* W_lr     = (const float*)d_in[5];   // [300,900]
  const float* W_o      = (const float*)d_in[6];   // [300,600]
  const float* b_o      = (const float*)d_in[7];
  const float* gru_bias = (const float*)d_in[8];
  const float* Wih_f    = (const float*)d_in[9];   // [900,300]
  const float* Whh_f    = (const float*)d_in[10];
  const float* bih_f    = (const float*)d_in[11];
  const float* bhh_f    = (const float*)d_in[12];
  const float* Wih_b    = (const float*)d_in[13];
  const float* Whh_b    = (const float*)d_in[14];
  const float* bih_b    = (const float*)d_in[15];
  const float* bhh_b    = (const float*)d_in[16];
  const int*   a2b      = (const int*)d_in[17];
  const int*   b2a      = (const int*)d_in[18];
  const int*   b2revb   = (const int*)d_in[19];
  float*       out      = (float*)d_out;

  // ---- workspace carve-out -----------------------------------------------
  char* base = (char*)d_ws;
  size_t off = 0;
  auto alloc = [&](size_t bytes) -> void* {
    void* p = base + off;
    off += (bytes + 255) & ~(size_t)255;
    return p;
  };

  uint16_t* bf_atoms  = (uint16_t*)alloc((size_t)A_ROWS * KF * 2);
  uint16_t* bf_bonds  = (uint16_t*)alloc((size_t)B_ROWS * KF * 2);
  uint16_t* wia_p     = (uint16_t*)alloc((size_t)HP * KF * 2);
  uint16_t* wib_p     = (uint16_t*)alloc((size_t)HP * KF * 2);
  uint16_t* wh_p      = (uint16_t*)alloc((size_t)5 * HP * HP * 2);
  uint16_t* wlr_p     = (uint16_t*)alloc((size_t)HP * (3 * HP) * 2);
  uint16_t* wihf_p    = (uint16_t*)alloc((size_t)960 * HP * 2);
  uint16_t* whhf_p    = (uint16_t*)alloc((size_t)960 * HP * 2);
  uint16_t* wihb_p    = (uint16_t*)alloc((size_t)960 * HP * 2);
  uint16_t* whhb_p    = (uint16_t*)alloc((size_t)960 * HP * 2);
  uint16_t* wo_p      = (uint16_t*)alloc((size_t)HP * (2 * HP) * 2);
  float*    input_atom   = (float*)alloc((size_t)A_ROWS * HID * 4);
  float*    input_bond   = (float*)alloc((size_t)B_ROWS * HID * 4);
  float*    message_atom = (float*)alloc((size_t)A_ROWS * HID * 4);
  float*    message_bond = (float*)alloc((size_t)B_ROWS * HID * 4);
  uint16_t* Xb           = (uint16_t*)alloc((size_t)B_ROWS * HP * 2);
  uint16_t* concat_b     = (uint16_t*)alloc((size_t)A_ROWS * (3 * HP) * 2);
  float*    node         = (float*)alloc((size_t)A_ROWS * HID * 4);
  uint16_t* msg_b        = (uint16_t*)alloc((size_t)A_ROWS * HP * 2);
  float*    gi_f         = (float*)alloc((size_t)(A_ROWS - 1) * GI_LD * 4);
  float*    gi_b         = (float*)alloc((size_t)(A_ROWS - 1) * GI_LD * 4);
  float*    gh_f         = (float*)alloc((size_t)NMOLS * GI_LD * 4);
  float*    gh_b         = (float*)alloc((size_t)NMOLS * GI_LD * 4);
  float*    h_f          = (float*)alloc((size_t)NMOLS * HID * 4);
  float*    h_b          = (float*)alloc((size_t)NMOLS * HID * 4);
  uint16_t* hbf_f        = (uint16_t*)alloc((size_t)NMOLS * HP * 2);
  uint16_t* hbf_b        = (uint16_t*)alloc((size_t)NMOLS * HP * 2);
  uint16_t* message_b    = (uint16_t*)alloc((size_t)A_ROWS * (2 * HP) * 2);
  float*    atom_hid     = (float*)alloc((size_t)A_ROWS * HID * 4);  // also agg temp

  const int TB = 256;
  #define L1D(total) dim3(cdiv((total), TB)), dim3(TB), 0, stream
  auto gemm = [&](const uint16_t* A, int lda, const uint16_t* W, int ldw,
                  float* C, int ldc, const float* resid, int ldr,
                  const float* bias, int M, int N, int K, int relu) {
    int Npad = ((N + 63) / 64) * 64;
    dim3 g(cdiv(M, 128), Npad / 64);
    hipLaunchKernelGGL(gemm_bf16_wmma, g, dim3(TB), 0, stream,
                       A, lda, W, ldw, C, ldc, resid, ldr, bias, M, N, K, relu);
  };

  // ---- 1. repack everything to padded bf16 -------------------------------
  hipLaunchKernelGGL(pack_bf16, L1D(A_ROWS * KF), f_atoms, 133, 133, bf_atoms, KF, KF, A_ROWS, A_ROWS);
  hipLaunchKernelGGL(pack_bf16, L1D(B_ROWS * KF), f_bonds, 147, 147, bf_bonds, KF, KF, B_ROWS, B_ROWS);
  hipLaunchKernelGGL(pack_bf16, L1D(HP * KF), W_i_atom, 133, 133, wia_p, KF, KF, HID, HP);
  hipLaunchKernelGGL(pack_bf16, L1D(HP * KF), W_i_bond, 147, 147, wib_p, KF, KF, HID, HP);
  for (int d = 0; d < 5; ++d)
    hipLaunchKernelGGL(pack_bf16, L1D(HP * HP), W_h + (size_t)d * HID * HID, HID, HID,
                       wh_p + (size_t)d * HP * HP, HP, HP, HID, HP);
  for (int c = 0; c < 3; ++c)   // W_lr: 900 K cols -> three 320 chunks
    hipLaunchKernelGGL(pack_bf16, L1D(HP * HP), W_lr + c * HID, 3 * HID, HID,
                       wlr_p + c * HP, 3 * HP, HP, HID, HP);
  hipLaunchKernelGGL(pack_bf16, L1D(960 * HP), Wih_f, HID, HID, wihf_p, HP, HP, 3 * HID, 960);
  hipLaunchKernelGGL(pack_bf16, L1D(960 * HP), Whh_f, HID, HID, whhf_p, HP, HP, 3 * HID, 960);
  hipLaunchKernelGGL(pack_bf16, L1D(960 * HP), Wih_b, HID, HID, wihb_p, HP, HP, 3 * HID, 960);
  hipLaunchKernelGGL(pack_bf16, L1D(960 * HP), Whh_b, HID, HID, whhb_p, HP, HP, 3 * HID, 960);
  for (int c = 0; c < 2; ++c)   // W_o: 600 K cols -> two 320 chunks
    hipLaunchKernelGGL(pack_bf16, L1D(HP * HP), W_o + c * HID, 2 * HID, HID,
                       wo_p + c * HP, 2 * HP, HP, HID, HP);

  // ---- 2. input projections (relu) ---------------------------------------
  gemm(bf_atoms, KF, wia_p, KF, input_atom, HID, nullptr, 0, nullptr, A_ROWS, HID, KF, 1);
  gemm(bf_bonds, KF, wib_p, KF, input_bond, HID, nullptr, 0, nullptr, B_ROWS, HID, KF, 1);
  hipMemcpyAsync(message_atom, input_atom, (size_t)A_ROWS * HID * 4, hipMemcpyDeviceToDevice, stream);
  hipMemcpyAsync(message_bond, input_bond, (size_t)B_ROWS * HID * 4, hipMemcpyDeviceToDevice, stream);

  // ---- 3. depth loop ------------------------------------------------------
  for (int d = 0; d < 5; ++d) {
    hipLaunchKernelGGL(agg_sum_max, L1D(A_ROWS * HID), message_bond, a2b, message_atom, 1);
    hipLaunchKernelGGL(bond_pre, L1D(B_ROWS * HP), message_atom, message_bond, b2a, b2revb, Xb);
    gemm(Xb, HP, wh_p + (size_t)d * HP * HP, HP, message_bond, HID,
         input_bond, HID, nullptr, B_ROWS, HID, HP, 1);
  }

  // ---- 4. final aggregation + concat GEMM (node = concat @ W_lr^T) -------
  hipLaunchKernelGGL(agg_sum_max, L1D(A_ROWS * HID), message_bond, a2b, atom_hid, 0);
  hipLaunchKernelGGL(pack_bf16, L1D(A_ROWS * HP), atom_hid,     HID, HID, concat_b,          3 * HP, HP, A_ROWS, A_ROWS);
  hipLaunchKernelGGL(pack_bf16, L1D(A_ROWS * HP), message_atom, HID, HID, concat_b + HP,     3 * HP, HP, A_ROWS, A_ROWS);
  hipLaunchKernelGGL(pack_bf16, L1D(A_ROWS * HP), input_atom,   HID, HID, concat_b + 2 * HP, 3 * HP, HP, A_ROWS, A_ROWS);
  gemm(concat_b, 3 * HP, wlr_p, 3 * HP, node, HID, nullptr, 0, nullptr, A_ROWS, HID, 3 * HP, 0);

  // ---- 5. BatchGRU --------------------------------------------------------
  hipLaunchKernelGGL(msg_make, L1D(A_ROWS * HP), node, gru_bias, msg_b);
  hipLaunchKernelGGL(fill_zero_u32, L1D((long long)A_ROWS * HP), (uint32_t*)message_b, (long long)A_ROWS * HP);
  hipLaunchKernelGGL(head_row, L1D(HID), node, gru_bias, message_b);
  hipLaunchKernelGGL(h0_init, L1D(NMOLS * HP), node, h_f, h_b, hbf_f, hbf_b);

  // gate precompute: gi = x @ Wih^T + bih (both directions, one big GEMM each)
  gemm(msg_b + HP, HP, wihf_p, HP, gi_f, GI_LD, nullptr, 0, bih_f, A_ROWS - 1, 3 * HID, HP, 0);
  gemm(msg_b + HP, HP, wihb_p, HP, gi_b, GI_LD, nullptr, 0, bih_b, A_ROWS - 1, 3 * HID, HP, 0);

  for (int t = 0; t < MOL; ++t) {
    // forward direction, time index t
    gemm(hbf_f, HP, whhf_p, HP, gh_f, GI_LD, nullptr, 0, bhh_f, NMOLS, 3 * HID, HP, 0);
    hipLaunchKernelGGL(gru_step, L1D(NMOLS * HID), gi_f, gh_f, h_f, hbf_f, message_b, t, 0);
    // backward direction, time index T-1-t
    gemm(hbf_b, HP, whhb_p, HP, gh_b, GI_LD, nullptr, 0, bhh_b, NMOLS, 3 * HID, HP, 0);
    hipLaunchKernelGGL(gru_step, L1D(NMOLS * HID), gi_b, gh_b, h_b, hbf_b, message_b, MOL - 1 - t, HP);
  }

  // ---- 6. output head -----------------------------------------------------
  gemm(message_b, 2 * HP, wo_p, 2 * HP, atom_hid, HID, nullptr, 0, b_o, A_ROWS, HID, 2 * HP, 1);
  hipLaunchKernelGGL(mol_mean, L1D(NMOLS * HID), atom_hid, out);
  #undef L1D
}
